// MoEGate_24996709663419
// MI455X (gfx1250) — matrix operand order repaired and checked by reading, not compiled, not run on hardware
//
#include <hip/hip_runtime.h>
#include <hip/hip_bf16.h>

typedef __attribute__((ext_vector_type(16))) __bf16 v16bf;
typedef __attribute__((ext_vector_type(8)))  float  v8f;
typedef unsigned int v4u __attribute__((ext_vector_type(4)));
typedef int          v8i __attribute__((ext_vector_type(8)));
typedef int          v4i __attribute__((ext_vector_type(4)));

#define HDIM   4096
#define NEXP   64
#define TOPK   8
#define SEQ    4096
#define NBATCH 4
#define TTOT   16384          // NBATCH * SEQ
#define TOK_PER_BLK 128       // 8 waves * 16 tokens
#define ALPHA_F 0.001f
#define NEG_INF (-3.4e38f)

#define KC     256            // K chunk staged in LDS per double-buffer slot
#define LROW   (KC + 8)       // padded LDS row in bf16 elems (528 B -> bank spread)
#define NCHUNK (HDIM / KC)    // 16

#if __has_builtin(__builtin_amdgcn_tensor_load_to_lds) && \
    __has_builtin(__builtin_amdgcn_s_wait_tensorcnt)
#define HAVE_TDM 1
#else
#define HAVE_TDM 0
#endif

// ---------------- weight fp32 -> bf16 (runs once per launch, 1 MB) ----------
__global__ void __launch_bounds__(256) wcvt_kernel(const float* __restrict__ w,
                                                   __bf16* __restrict__ o) {
    int i = (blockIdx.x * 256 + threadIdx.x) * 4;
    float4 v = *(const float4*)(w + i);
    o[i + 0] = (__bf16)v.x;
    o[i + 1] = (__bf16)v.y;
    o[i + 2] = (__bf16)v.z;
    o[i + 3] = (__bf16)v.w;
}

#if HAVE_TDM
// Issue one TDM 2D-tile load: 8 expert rows x KC bf16, global->LDS,
// with pad_amount=4 dwords every 128 dwords (= +16 B per row) so the LDS
// layout lands on the padded LROW stride.
__device__ __forceinline__ void tdm_issue(const __bf16* gsrc, void* ldst) {
    unsigned long long ga = (unsigned long long)(uintptr_t)gsrc;
    unsigned int lo = (unsigned int)(uintptr_t)ldst;   // low 32b = LDS offset
    v4u g0;
    g0[0] = 1u;                                        // count=1 (valid user D#)
    g0[1] = lo;                                        // lds_addr
    g0[2] = (unsigned)(ga & 0xffffffffu);              // global_addr[31:0]
    g0[3] = (unsigned)((ga >> 32) & 0x1ffffffu) | (2u << 30); // [56:32] | type=2
    v8i g1;
    g1[0] = (1 << 16)     // data_size = 2 bytes
          | (1 << 20)     // pad_enable
          | (6 << 22)     // pad_interval: 128 dwords (one 512 B row)
          | (3 << 25);    // pad_amount: 4 dwords (16 B)
    g1[1] = (int)((unsigned)HDIM << 16);   // tensor_dim0 = 4096 (bits 79:48)
    g1[2] = (int)((unsigned)NEXP << 16);   // tensor_dim1 = 64   (bits 111:80)
    g1[3] = (int)((unsigned)KC   << 16);   // tile_dim0  = 256   (bits 127:112)
    g1[4] = 8;                             // tile_dim1  = 8 expert rows / wave
    g1[5] = HDIM;                          // tensor_dim0_stride = 4096 elems
    g1[6] = 0;
    g1[7] = 0;
    v4i z4 = {0, 0, 0, 0};
    v8i z8 = {0, 0, 0, 0, 0, 0, 0, 0};
    // clang-23 / therock lane: 6-arg form (extra int32x8 descriptor group)
    __builtin_amdgcn_tensor_load_to_lds(g0, g1, z4, z4, z8, 0);
}
#endif

// ---------------- main gate kernel ------------------------------------------
// block = 256 threads = 8 waves; wave w owns tokens [TB + 16w, TB + 16w + 16)
__global__ void __launch_bounds__(256) moe_gate_kernel(
        const float*  __restrict__ x,      // [TTOT, HDIM] fp32
        const __bf16* __restrict__ wbf,    // [NEXP, HDIM] bf16
        const float*  __restrict__ bias,   // [NEXP]
        float*        __restrict__ out,    // idx | weight | aux
        float*        __restrict__ gCnt,   // [NBATCH, NEXP]
        float*        __restrict__ gSum) { // [NBATCH, NEXP]
    // B staging and score scratch are never live at the same time -> union
    __shared__ union {
        __bf16 B[2][NEXP][LROW];           // 67,584 B double-buffered weights
        float  sc[8][16][72];              // 36,864 B sigmoid scores
    } u;
    __shared__ float sBias[NEXP];
    __shared__ float sCnt[NEXP];
    __shared__ float sSum[NEXP];

    const int tid  = threadIdx.x;
    const int wv   = tid >> 5;
    const int lane = tid & 31;
    const int half = lane >> 4;            // 0: lanes 0-15, 1: lanes 16-31
    const int col  = lane & 15;
    const int TB   = blockIdx.x * TOK_PER_BLK;        // first token of block
    const int batch = TB / SEQ;                       // block never straddles

    if (tid < NEXP) {
        sBias[tid] = bias[tid];
        sCnt[tid]  = 0.0f;
        sSum[tid]  = 0.0f;
    }

    // ---- GEMM: 16 tokens x 64 experts, K = HDIM via bf16 WMMA --------------
    const int tok = TB + wv * 16 + col;               // A row for this lane
    const float* pA = x + (size_t)tok * HDIM + half * 16;

    v8f acc[4];
#pragma unroll
    for (int nt = 0; nt < 4; ++nt) acc[nt] = (v8f){0.f,0.f,0.f,0.f,0.f,0.f,0.f,0.f};

#if HAVE_TDM
    // prologue: every wave DMAs its 8-expert slice of chunk 0 into buffer 0
    tdm_issue(wbf + (size_t)(wv * 8) * HDIM, &u.B[0][wv * 8][0]);
#endif

    for (int c = 0; c < NCHUNK; ++c) {
        const int kb = c * KC;
#if HAVE_TDM
        if (c < NCHUNK - 1) {   // prefetch next chunk, then wait for current
            tdm_issue(wbf + (size_t)(wv * 8) * HDIM + (kb + KC),
                      &u.B[(c + 1) & 1][wv * 8][0]);
            __builtin_amdgcn_s_wait_tensorcnt(1);
        } else {
            __builtin_amdgcn_s_wait_tensorcnt(0);
        }
#else
        {   // fallback: cooperative global->LDS copy of chunk c
            const __bf16* src = wbf + kb;
            for (int i = tid; i < (NEXP * KC) / 8; i += 256) {
                int row = (i * 8) / KC, kk = (i * 8) % KC;
                *(uint4*)&u.B[c & 1][row][kk] =
                    *(const uint4*)(src + (size_t)row * HDIM + kk);
            }
        }
#endif
        __syncthreads();        // chunk c visible to all waves

        const __bf16* lB = &u.B[c & 1][0][0];
#pragma unroll 2
        for (int kl = 0; kl < KC; kl += 32) {
            const int k0 = kb + kl;
            // A fragment: 16 fp32 -> bf16 (K {0..15} / {16..31})
            float4 a0 = *(const float4*)(pA + k0);
            float4 a1 = *(const float4*)(pA + k0 + 4);
            float4 a2 = *(const float4*)(pA + k0 + 8);
            float4 a3 = *(const float4*)(pA + k0 + 12);
            v16bf af;
            af[0]=(__bf16)a0.x; af[1]=(__bf16)a0.y; af[2]=(__bf16)a0.z; af[3]=(__bf16)a0.w;
            af[4]=(__bf16)a1.x; af[5]=(__bf16)a1.y; af[6]=(__bf16)a1.z; af[7]=(__bf16)a1.w;
            af[8]=(__bf16)a2.x; af[9]=(__bf16)a2.y; af[10]=(__bf16)a2.z; af[11]=(__bf16)a2.w;
            af[12]=(__bf16)a3.x; af[13]=(__bf16)a3.y; af[14]=(__bf16)a3.z; af[15]=(__bf16)a3.w;

#pragma unroll
            for (int nt = 0; nt < 4; ++nt) {
                // B fragment from LDS: 16 contiguous bf16 K-values
                v16bf bf = *(const v16bf*)(lB + (size_t)(nt * 16 + col) * LROW
                                              + kl + half * 16);
                acc[nt] = __builtin_amdgcn_wmma_f32_16x16x32_bf16(
                              false, af, false, bf, (short)0, acc[nt], false, false);
            }
        }
        __syncthreads();        // chunk c consumed before its buffer is reused
    }

    // ---- sigmoid, stash scores in LDS (overlays B buffers), expert sums ----
#pragma unroll
    for (int nt = 0; nt < 4; ++nt) {
        float rs = 0.0f;
#pragma unroll
        for (int r = 0; r < 8; ++r) {
            float s = 1.0f / (1.0f + __expf(-acc[nt][r]));
            u.sc[wv][r + 8 * half][nt * 16 + col] = s;
            rs += s;
        }
        atomicAdd(&sSum[nt * 16 + col], rs);
    }
    __syncthreads();

    // ---- top-8 selection: one thread per token (threads 0..127) ------------
    if (tid < TOK_PER_BLK) {
        const int w2 = tid >> 4, m = tid & 15;
        const int t  = TB + tid;

        float arr[NEXP];
#pragma unroll
        for (int c = 0; c < 16; ++c) {
            float4 v = *(const float4*)&u.sc[w2][m][c * 4];
            arr[c*4+0] = v.x + sBias[c*4+0];
            arr[c*4+1] = v.y + sBias[c*4+1];
            arr[c*4+2] = v.z + sBias[c*4+2];
            arr[c*4+3] = v.w + sBias[c*4+3];
        }

        int   idxs[TOPK];
        float wts[TOPK];
        float denom = 1e-20f;
#pragma unroll
        for (int j = 0; j < TOPK; ++j) {
            float best = NEG_INF; int bi = 0;
#pragma unroll
            for (int e = 0; e < NEXP; ++e)
                if (arr[e] > best) { best = arr[e]; bi = e; }
            idxs[j] = bi;
            wts[j]  = u.sc[w2][m][bi];      // raw (unbiased) sigmoid score
            denom  += wts[j];
#pragma unroll
            for (int e = 0; e < NEXP; ++e)
                if (e == bi) arr[e] = NEG_INF;
        }

        const float scl = 2.5f / denom;
#pragma unroll
        for (int j = 0; j < TOPK; ++j) {
            out[(size_t)t * TOPK + j]                       = (float)idxs[j];
            out[(size_t)TTOT * TOPK + (size_t)t * TOPK + j] = wts[j] * scl;
            atomicAdd(&sCnt[idxs[j]], 1.0f);
        }
    }
    __syncthreads();

    // ---- flush block accumulators ------------------------------------------
    if (tid < NEXP) {
        atomicAdd(&gCnt[batch * NEXP + tid], sCnt[tid]);
        atomicAdd(&gSum[batch * NEXP + tid], sSum[tid]);
    }
}

// ---------------- aux loss reduction ----------------------------------------
__global__ void __launch_bounds__(64) aux_loss_kernel(const float* __restrict__ gCnt,
                                                      const float* __restrict__ gSum,
                                                      float* __restrict__ outAux) {
    __shared__ float red[NEXP];
    const int e = threadIdx.x;
    float p = 0.0f;
#pragma unroll
    for (int b = 0; b < NBATCH; ++b)
        p += gCnt[b * NEXP + e] * gSum[b * NEXP + e];
    red[e] = p;
    __syncthreads();
    if (e == 0) {
        float tot = 0.0f;
#pragma unroll
        for (int i = 0; i < NEXP; ++i) tot += red[i];
        // ce = cnt * E/(S*K); mean_scores = sum/S; mean over B; * alpha
        const float c = ALPHA_F * ((float)NEXP / ((float)SEQ * (float)TOPK))
                        / (float)SEQ / (float)NBATCH;
        outAux[0] = tot * c;
    }
}

// ---------------- host launcher ---------------------------------------------
extern "C" void kernel_launch(void* const* d_in, const int* in_sizes, int n_in,
                              void* d_out, int out_size, void* d_ws, size_t ws_size,
                              hipStream_t stream) {
    const float* x    = (const float*)d_in[0];   // [4,4096,4096]
    const float* wt   = (const float*)d_in[1];   // [64,4096]
    const float* bias = (const float*)d_in[2];   // [1,64]
    float* out = (float*)d_out;

    float*  gCnt = (float*)d_ws;                          // 256 floats
    float*  gSum = (float*)d_ws + NBATCH * NEXP;          // 256 floats
    __bf16* wbf  = (__bf16*)((char*)d_ws + 2048);         // 512 KB bf16 weight

    (void)hipMemsetAsync(d_ws, 0, 2048, stream);          // zero accumulators

    wcvt_kernel<<<(NEXP * HDIM) / (256 * 4), 256, 0, stream>>>(wt, wbf);

    moe_gate_kernel<<<TTOT / TOK_PER_BLK, 256, 0, stream>>>(
        x, wbf, bias, out, gCnt, gSum);

    aux_loss_kernel<<<1, 64, 0, stream>>>(gCnt, gSum,
        out + (size_t)2 * TTOT * TOPK);
}
